// GAT_20873541058759
// MI455X (gfx1250) — matrix operand order repaired and checked by reading, not compiled, not run on hardware
//
#include <hip/hip_runtime.h>

typedef __attribute__((ext_vector_type(16))) __bf16 v16bf;
typedef __attribute__((ext_vector_type(8)))  __bf16 v8bf;
typedef __attribute__((ext_vector_type(8)))  float  v8f;

#define HEADS 8
#define HID 32
#define C1 256            // HEADS*HID
#define NEG_SLOPE 0.2f

static __device__ __forceinline__ __bf16 f2bf(float f) {
    union { float f; unsigned u; } x; x.f = f;
    unsigned r = (x.u + 0x7FFFu + ((x.u >> 16) & 1u)) >> 16;   // RNE truncate
    union { unsigned short s; __bf16 b; } y; y.s = (unsigned short)r;
    return y.b;
}

static __device__ __forceinline__ void atomicMaxF(float* addr, float v) {
    // monotone trick: signed max for non-negatives, unsigned min for negatives
    if (v >= 0.f) atomicMax((int*)addr, __float_as_int(v));
    else          atomicMin((unsigned int*)addr, __float_as_uint(v));
}

// ---------------- f32 -> bf16 cast (4 elems/thread)
__global__ void cast_bf16(const float* __restrict__ in, __bf16* __restrict__ outb, long n4) {
    long i = ((long)blockIdx.x * blockDim.x + threadIdx.x);
    if (i >= n4) return;
    float4 f = *(const float4*)(in + i * 4);
    __bf16* o = outb + i * 4;
    o[0] = f2bf(f.x); o[1] = f2bf(f.y); o[2] = f2bf(f.z); o[3] = f2bf(f.w);
}

// ---------------- weight convert + transpose: W[K x Nout] f32 -> Wt[Nout x K] bf16
__global__ void w_to_bf16t(const float* __restrict__ W, __bf16* __restrict__ Wt,
                           int Kdim, int Nout) {
    int idx = blockIdx.x * blockDim.x + threadIdx.x;
    if (idx >= Kdim * Nout) return;
    int k = idx / Nout, n = idx % Nout;
    Wt[(long)n * Kdim + k] = f2bf(W[idx]);
}

// ---------------- WMMA GEMM: C[M x Nout] = A[M x K] bf16 * Bt^T (Bt: [Nout x K] bf16)
// One wave computes a 16x64 slab of C: one A fragment reused across 4 WMMAs per K-step.
__global__ void gemm_bf16(const __bf16* __restrict__ A, const __bf16* __restrict__ Bt,
                          float* __restrict__ C, int M, int K, int Nout) {
    int lane = threadIdx.x & 31;
    int wave = threadIdx.x >> 5;
    int ntn  = Nout >> 6;                 // 64-wide column groups
    long ntm = (M + 15) >> 4;
    long slab = (long)blockIdx.x * 4 + wave;
    if (slab >= ntm * (long)ntn) return;
    int mt = (int)(slab / ntn);
    int ng = (int)(slab % ntn);
    int half = lane >> 4;                 // half-wave
    int l16  = lane & 15;
    int m  = mt * 16 + l16;
    int mc = m < M ? m : (M - 1);         // clamp loads; stores are guarded
    // A 16-bit 16x32 fragment: lane half h needs K = k + g*16 + h*8 + [0..7]
    const __bf16* Arow = A + (long)mc * K + half * 8;
    // B fragment: lane half h holds K = k + g*16 + h*8 + [0..7] of a column (same as A)
    const __bf16* B0 = Bt + (long)(ng * 64 + l16) * K + half * 8;

    v8f acc[4] = {};
    for (int k = 0; k < K; k += 32) {
        v16bf a;
        v8bf a0 = *(const v8bf*)(Arow + k);
        v8bf a1 = *(const v8bf*)(Arow + k + 16);
        #pragma unroll
        for (int i = 0; i < 8; ++i) { a[i] = a0[i]; a[8 + i] = a1[i]; }
        #pragma unroll
        for (int t4 = 0; t4 < 4; ++t4) {
            const __bf16* Bp = B0 + (long)(t4 * 16) * K + k;
            v8bf b0 = *(const v8bf*)Bp;
            v8bf b1 = *(const v8bf*)(Bp + 16);
            v16bf b;
            #pragma unroll
            for (int i = 0; i < 8; ++i) { b[i] = b0[i]; b[8 + i] = b1[i]; }
            acc[t4] = __builtin_amdgcn_wmma_f32_16x16x32_bf16(
                false, a, false, b, (short)0, acc[t4], false, false);
        }
    }
    // C/D layout: vgpr r -> row mt*16 + half*8 + r
    #pragma unroll
    for (int t4 = 0; t4 < 4; ++t4) {
        int ncol = ng * 64 + t4 * 16 + l16;
        #pragma unroll
        for (int r = 0; r < 8; ++r) {
            int row = mt * 16 + half * 8 + r;
            if (row < M) C[(long)row * Nout + ncol] = acc[t4][r];
        }
    }
}

// ---------------- per-node attention projections el/er
__global__ void node_dots(const float* __restrict__ feat, const float* __restrict__ al,
                          const float* __restrict__ ar, float* __restrict__ el,
                          float* __restrict__ er, int N) {
    int t = blockIdx.x * blockDim.x + threadIdx.x;
    if (t >= N * HEADS) return;
    int nn = t >> 3, h = t & 7;
    const float* f = feat + (long)nn * C1 + h * HID;
    const float* a = al + h * HID;
    const float* b = ar + h * HID;
    float sl = 0.f, sr = 0.f;
    #pragma unroll
    for (int i = 0; i < HID; ++i) { float v = f[i]; sl += v * a[i]; sr += v * b[i]; }
    el[t] = sl; er[t] = sr;
}

// ---------------- per-layer accumulator init
__global__ void init_layer(float* __restrict__ rst, float* __restrict__ maxv,
                           float* __restrict__ denom, int N) {
    long idx = (long)blockIdx.x * blockDim.x + threadIdx.x;
    if (idx < (long)N * C1) rst[idx] = 0.f;
    if (idx < (long)N * HEADS) { maxv[idx] = -3.0e38f; denom[idx] = 0.f; }
}

static __device__ __forceinline__ float edge_logit(const float* el, const float* er,
                                                   int s, int d, int h) {
    float v = el[(long)s * HEADS + h] + er[(long)d * HEADS + h];
    return v > 0.f ? v : NEG_SLOPE * v;
}

__global__ void edge_max_k(const int* __restrict__ src, const int* __restrict__ dst,
                           const float* __restrict__ el, const float* __restrict__ er,
                           float* __restrict__ maxv, long EH) {
    long t = (long)blockIdx.x * blockDim.x + threadIdx.x;
    if (t >= EH) return;
    long e = t >> 3; int h = (int)(t & 7);
    int s = src[e], d = dst[e];
    atomicMaxF(&maxv[(long)d * HEADS + h], edge_logit(el, er, s, d, h));
}

__global__ void edge_sum_k(const int* __restrict__ src, const int* __restrict__ dst,
                           const float* __restrict__ el, const float* __restrict__ er,
                           const float* __restrict__ maxv, float* __restrict__ denom,
                           long EH) {
    long t = (long)blockIdx.x * blockDim.x + threadIdx.x;
    if (t >= EH) return;
    long e = t >> 3; int h = (int)(t & 7);
    int s = src[e], d = dst[e];
    long di = (long)d * HEADS + h;
    unsafeAtomicAdd(&denom[di], __expf(edge_logit(el, er, s, d, h) - maxv[di]));
}

__global__ void edge_scatter_k(const int* __restrict__ src, const int* __restrict__ dst,
                               const float* __restrict__ el, const float* __restrict__ er,
                               const float* __restrict__ maxv, const float* __restrict__ denom,
                               const float* __restrict__ feat, float* __restrict__ rst,
                               long EH) {
    long t = (long)blockIdx.x * blockDim.x + threadIdx.x;
    if (t >= EH) return;
    long e = t >> 3; int h = (int)(t & 7);
    int s = src[e], d = dst[e];
    const float* fs = feat + (long)s * C1 + h * HID;
    __builtin_prefetch(fs, 0, 1);            // global_prefetch: warm the gather row
    long di = (long)d * HEADS + h;
    float dn = denom[di];
    float alpha = __expf(edge_logit(el, er, s, d, h) - maxv[di]) / (dn == 0.f ? 1.f : dn);
    float* rd = rst + (long)d * C1 + h * HID;
    #pragma unroll
    for (int i = 0; i < HID; ++i) unsafeAtomicAdd(rd + i, alpha * fs[i]);
}

// ---------------- (+residual) + bias, elu; optional fused bf16 copy for next GEMM
__global__ void finalize_k(const float* __restrict__ rst, const float* __restrict__ resid,
                           const float* __restrict__ bias, float* __restrict__ outh,
                           __bf16* __restrict__ outb, long total) {
    long idx = (long)blockIdx.x * blockDim.x + threadIdx.x;
    if (idx >= total) return;
    int c = (int)(idx % C1);
    float v = rst[idx] + bias[c];
    if (resid) v += resid[idx];
    float r = v > 0.f ? v : __expf(v) - 1.f;   // elu
    outh[idx] = r;
    if (outb) outb[idx] = f2bf(r);
}

// ---------------- column sums of h2 (for mean-then-linear readout)
__global__ void zero_k(float* __restrict__ p, int n) {
    int i = blockIdx.x * blockDim.x + threadIdx.x;
    if (i < n) p[i] = 0.f;
}

__global__ void colsum_k(const float* __restrict__ h, float* __restrict__ colsum, int N) {
    int t = threadIdx.x;                  // 0..255 = column
    float s = 0.f;
    for (int r = blockIdx.x; r < N; r += gridDim.x)
        s += h[(long)r * C1 + t];
    unsafeAtomicAdd(&colsum[t], s);
}

__global__ void final_k(const float* __restrict__ colsum, const float* __restrict__ Wl,
                        const float* __restrict__ bl, float* __restrict__ out,
                        int N, int Kdim, int Nout) {
    int j = blockIdx.x * blockDim.x + threadIdx.x;
    if (j >= Nout) return;
    float inv = 1.f / (float)N;
    float s = bl[j];
    for (int k = 0; k < Kdim; ++k) s += (colsum[k] * inv) * Wl[(long)k * Nout + j];
    out[j] = s;
}

extern "C" void kernel_launch(void* const* d_in, const int* in_sizes, int n_in,
                              void* d_out, int out_size, void* d_ws, size_t ws_size,
                              hipStream_t stream) {
    const float* x   = (const float*)d_in[0];
    const int*   src = (const int*)d_in[1];
    const int*   dst = (const int*)d_in[2];
    const float* W1  = (const float*)d_in[3];
    const float* al1 = (const float*)d_in[4];
    const float* ar1 = (const float*)d_in[5];
    const float* b1  = (const float*)d_in[6];
    const float* W2  = (const float*)d_in[7];
    const float* al2 = (const float*)d_in[8];
    const float* ar2 = (const float*)d_in[9];
    const float* b2  = (const float*)d_in[10];
    const float* Wl  = (const float*)d_in[11];
    const float* bl  = (const float*)d_in[12];
    float* out = (float*)d_out;

    const int IN = 128, OUT = 128;
    int N = in_sizes[0] / IN;
    int E = in_sizes[1];

    // ---- workspace carve-up (~200 MB)
    char* w = (char*)d_ws;
    auto alloc = [&](size_t bytes) -> char* {
        char* p = w; w += (bytes + 255) & ~(size_t)255; return p;
    };
    float*  feat   = (float*)alloc((size_t)N * C1 * 4);    // feat1, reused as feat2
    float*  h1     = (float*)alloc((size_t)N * C1 * 4);    // rst1 -> h1
    float*  h2     = (float*)alloc((size_t)N * C1 * 4);    // rst2 -> h2
    float*  el     = (float*)alloc((size_t)N * HEADS * 4);
    float*  er     = (float*)alloc((size_t)N * HEADS * 4);
    float*  maxv   = (float*)alloc((size_t)N * HEADS * 4);
    float*  denom  = (float*)alloc((size_t)N * HEADS * 4);
    float*  colsum = (float*)alloc(C1 * 4);
    __bf16* xb     = (__bf16*)alloc((size_t)N * IN * 2);   // bf16 activations for GEMM1
    __bf16* h1b    = (__bf16*)alloc((size_t)N * C1 * 2);   // bf16 activations for GEMM2
    __bf16* w1t    = (__bf16*)alloc((size_t)IN * C1 * 2);
    __bf16* w2t    = (__bf16*)alloc((size_t)C1 * C1 * 2);
    (void)ws_size; (void)n_in; (void)out_size;

    // ---- weights -> bf16 transposed; x -> bf16
    w_to_bf16t<<<(IN * C1 + 255) / 256, 256, 0, stream>>>(W1, w1t, IN, C1);
    w_to_bf16t<<<(C1 * C1 + 255) / 256, 256, 0, stream>>>(W2, w2t, C1, C1);
    long xn4 = (long)N * IN / 4;
    cast_bf16<<<(unsigned)((xn4 + 255) / 256), 256, 0, stream>>>(x, xb, xn4);

    long tot = (long)N * C1;
    unsigned ngrid = (unsigned)((tot + 255) / 256);
    int NH = N * HEADS;
    long EH = (long)E * HEADS;
    unsigned egrid = (unsigned)((EH + 255) / 256);
    long slabs = ((long)((N + 15) / 16)) * (C1 / 64);      // 16x64 per wave
    unsigned ggrid = (unsigned)((slabs + 3) / 4);

    // ---- layer 1
    gemm_bf16<<<ggrid, 128, 0, stream>>>(xb, w1t, feat, N, IN, C1);
    node_dots<<<(NH + 255) / 256, 256, 0, stream>>>(feat, al1, ar1, el, er, N);
    init_layer<<<ngrid, 256, 0, stream>>>(h1, maxv, denom, N);
    edge_max_k<<<egrid, 256, 0, stream>>>(src, dst, el, er, maxv, EH);
    edge_sum_k<<<egrid, 256, 0, stream>>>(src, dst, el, er, maxv, denom, EH);
    edge_scatter_k<<<egrid, 256, 0, stream>>>(src, dst, el, er, maxv, denom, feat, h1, EH);
    finalize_k<<<ngrid, 256, 0, stream>>>(h1, nullptr, b1, h1, h1b, tot);

    // ---- layer 2 (residual)
    gemm_bf16<<<ggrid, 128, 0, stream>>>(h1b, w2t, feat, N, C1, C1);
    node_dots<<<(NH + 255) / 256, 256, 0, stream>>>(feat, al2, ar2, el, er, N);
    init_layer<<<ngrid, 256, 0, stream>>>(h2, maxv, denom, N);
    edge_max_k<<<egrid, 256, 0, stream>>>(src, dst, el, er, maxv, EH);
    edge_sum_k<<<egrid, 256, 0, stream>>>(src, dst, el, er, maxv, denom, EH);
    edge_scatter_k<<<egrid, 256, 0, stream>>>(src, dst, el, er, maxv, denom, feat, h2, EH);
    finalize_k<<<ngrid, 256, 0, stream>>>(h2, h1, b2, h2, (__bf16*)nullptr, tot);

    // ---- readout: mean(h2) @ Wl + bl
    zero_k<<<1, 256, 0, stream>>>(colsum, C1);
    colsum_k<<<256, 256, 0, stream>>>(h2, colsum, N);
    final_k<<<1, 128, 0, stream>>>(colsum, Wl, bl, out, N, C1, OUT);
}
